// Decoder_67302137528584
// MI455X (gfx1250) — matrix operand order repaired and checked by reading, not compiled, not run on hardware
//
#include <hip/hip_runtime.h>
#include <math.h>

// ---------------- problem constants ----------------
#define BB   64
#define PP   400
#define PPAD 416      // P padded to multiple of 32 for WMMA K
#define ED   512
#define DD   512
#define EMB  256
#define AD   256
#define CD   256
#define VV   32000
#define TT   32

// ---------------- WMMA types ----------------
typedef __attribute__((ext_vector_type(16))) __bf16 v16bf;
typedef __attribute__((ext_vector_type(8)))  float  v8f;
typedef __attribute__((ext_vector_type(4)))  unsigned int v4u;

union ABFrag { v16bf bf; v4u u[2]; };

__device__ __forceinline__ unsigned short f2bf(float f) {
    unsigned int u = __float_as_uint(f);
    unsigned int r = u + 0x7FFFu + ((u >> 16) & 1u);   // round-to-nearest-even
    return (unsigned short)(r >> 16);
}

// ---------------- generic per-tile bf16 WMMA GEMM (used for att1, M=25600) ----------------
// C[M][N] = A[M][K] * B[N][K]^T + bias. One wave per 16x16 tile, 8 waves/block.
__global__ __launch_bounds__(256) void gemm_bf16_nt(
    const unsigned short* __restrict__ A,
    const unsigned short* __restrict__ B,
    const float* __restrict__ bias,
    float* __restrict__ C,
    int M, int N, int K)
{
    int wave = threadIdx.x >> 5;
    int lane = threadIdx.x & 31;
    int tile = blockIdx.x * (blockDim.x >> 5) + wave;
    int ntn  = N >> 4;
    int tm   = tile / ntn;
    int tn   = tile - tm * ntn;
    if (tm >= (M >> 4)) return;

    int r  = lane & 15;
    int kh = lane >> 4;
    const unsigned short* ap = A + (size_t)(tm * 16 + r) * K + 8 * kh;   // A: K {0..7,16..23}/{8..15,24..31}
    const unsigned short* bp = B + (size_t)(tn * 16 + r) * K + 8 * kh;   // B: K {0..7,16..23}/{8..15,24..31}

    v8f acc = {0.f, 0.f, 0.f, 0.f, 0.f, 0.f, 0.f, 0.f};
    for (int k = 0; k < K; k += 32) {
        ABFrag a, b;
        a.u[0] = *(const v4u*)(ap + k);
        a.u[1] = *(const v4u*)(ap + k + 16);
        b.u[0] = *(const v4u*)(bp + k);
        b.u[1] = *(const v4u*)(bp + k + 16);
        acc = __builtin_amdgcn_wmma_f32_16x16x32_bf16(
            false, a.bf, false, b.bf, (short)0, acc, false, false);
    }

    int n  = tn * 16 + r;
    float bv = bias ? bias[n] : 0.0f;
    int mbase = tm * 16 + (kh << 3);
    #pragma unroll
    for (int v = 0; v < 8; ++v)
        C[(size_t)(mbase + v) * N + n] = acc[v] + bv;
}

// ---------------- M=64 strip GEMM with async-to-LDS A staging ----------------
// C[64][N] = A[64][K] * B[N][K]^T + bias. Each wave computes a 64x16 strip
// (4 stacked 16x16 tiles): B fragment loaded once, reused by 4 WMMAs.
// A (<=64KB) staged to LDS by the whole block via GLOBAL_LOAD_ASYNC_TO_LDS_B128,
// then read as fragments with ds_load_b128. Rows padded +8 elems (16B) so the
// 16 fragment lanes hit distinct LDS banks (row stride 260 dwords).
__global__ __launch_bounds__(256) void gemm_bf16_strip(
    const unsigned short* __restrict__ A,   // [64][K]
    const unsigned short* __restrict__ B,   // [N][K]
    const float* __restrict__ bias,
    float* __restrict__ C,                  // [64][N]
    int N, int K)
{
    extern __shared__ __align__(16) unsigned short shA[];   // [64][K+8]
    const int KP = K + 8;
    int tid = threadIdx.x;

    // ---- stage A into LDS (async copy, ASYNCcnt-tracked) ----
    int cpr    = K >> 3;                    // 16B chunks per row
    int chunks = 64 * cpr;
    for (int c = tid; c < chunks; c += 256) {
        int row = c / cpr;
        int col = (c - row * cpr) << 3;
        const unsigned short* g = A + (size_t)row * K + col;
        unsigned lds = (unsigned)(size_t)(&shA[row * KP + col]);
        asm volatile("global_load_async_to_lds_b128 %0, %1, off"
                     :: "v"(lds), "v"(g) : "memory");
    }
    asm volatile("s_wait_asynccnt 0x0" ::: "memory");
    __syncthreads();

    int wave = tid >> 5, lane = tid & 31;
    int tn = blockIdx.x * 8 + wave;
    if (tn >= (N >> 4)) return;

    int r  = lane & 15;
    int kh = lane >> 4;
    const unsigned short* bp = B + (size_t)(tn * 16 + r) * K + 8 * kh;

    v8f acc[4];
    #pragma unroll
    for (int mt = 0; mt < 4; ++mt)
        acc[mt] = (v8f){0.f, 0.f, 0.f, 0.f, 0.f, 0.f, 0.f, 0.f};

    for (int k = 0; k < K; k += 32) {
        ABFrag b;
        b.u[0] = *(const v4u*)(bp + k);
        b.u[1] = *(const v4u*)(bp + k + 16);
        #pragma unroll
        for (int mt = 0; mt < 4; ++mt) {
            const unsigned short* ap = &shA[(mt * 16 + r) * KP + 8 * kh + k];
            ABFrag a;
            a.u[0] = *(const v4u*)(ap);
            a.u[1] = *(const v4u*)(ap + 16);
            acc[mt] = __builtin_amdgcn_wmma_f32_16x16x32_bf16(
                false, a.bf, false, b.bf, (short)0, acc[mt], false, false);
        }
    }

    int n  = tn * 16 + r;
    float bv = bias ? bias[n] : 0.0f;
    #pragma unroll
    for (int mt = 0; mt < 4; ++mt) {
        int mbase = mt * 16 + (kh << 3);
        #pragma unroll
        for (int v = 0; v < 8; ++v)
            C[(size_t)(mbase + v) * N + n] = acc[mt][v] + bv;
    }
}

// ---------------- helpers ----------------
__global__ void k_f2bf(const float* __restrict__ s, unsigned short* __restrict__ d, int n) {
    int i = blockIdx.x * 256 + threadIdx.x;
    if (i < n) d[i] = f2bf(s[i]);
}

__global__ void k_zero(float* p, int n) {
    int i = blockIdx.x * 256 + threadIdx.x;
    if (i < n) p[i] = 0.f;
}

// conv_w [CD][PP] -> bf16 padded [CD][PPAD]
__global__ void k_convw_bf(const float* __restrict__ s, unsigned short* __restrict__ d) {
    int i = blockIdx.x * 256 + threadIdx.x;          // CD*PPAD
    int row = i / PPAD, col = i - row * PPAD;
    d[i] = (col < PP) ? f2bf(s[row * PP + col]) : (unsigned short)0;
}

// stable descending argsort of lengths; also emit caps/lens/sort_ind outputs (as float)
__global__ void k_sort(const int* __restrict__ lens, const int* __restrict__ caps,
                       int* sort_ind, int* lens_s, int* caps_s,
                       float* out_caps, float* out_lens, float* out_sortind) {
    int i = threadIdx.x;                             // 64 threads
    __shared__ int sl[BB];
    sl[i] = lens[i];
    __syncthreads();
    int li = sl[i], rk = 0;
    for (int j = 0; j < BB; ++j) {
        int lj = sl[j];
        if (lj > li || (lj == li && j < i)) rk++;
    }
    sort_ind[rk] = i;
    lens_s[rk]   = li;
    out_lens[rk]    = (float)li;
    out_sortind[rk] = (float)i;
    for (int j = 0; j <= TT; ++j) {
        int c = caps[i * (TT + 1) + j];
        caps_s[rk * (TT + 1) + j] = c;
        out_caps[rk * (TT + 1) + j] = (float)c;
    }
}

// gather enc by sort_ind and convert to bf16: dst[b][p][d]
__global__ void k_gather_enc_bf(const float* __restrict__ enc, const int* __restrict__ si,
                                unsigned short* __restrict__ dst) {
    size_t i = (size_t)blockIdx.x * 256 + threadIdx.x;  // BB*PP*ED
    int b = (int)(i / (PP * ED));
    size_t rem = i - (size_t)b * (PP * ED);
    dst[i] = f2bf(enc[(size_t)si[b] * PP * ED + rem]);
}

// meanenc[b][d] = mean over p of enc_sorted
__global__ void k_mean_enc(const float* __restrict__ enc, const int* __restrict__ si,
                           float* __restrict__ m) {
    int i = blockIdx.x * 256 + threadIdx.x;          // BB*ED
    int b = i >> 9, d = i & (ED - 1);
    const float* e = enc + (size_t)si[b] * PP * ED + d;
    float s = 0.f;
    for (int p = 0; p < PP; ++p) s += e[(size_t)p * ED];
    m[i] = s * (1.0f / PP);
}

__global__ void k_embed0(const int* __restrict__ caps_s, const float* __restrict__ embd,
                         float* __restrict__ emb) {
    int b = blockIdx.x, t = threadIdx.x;             // 64 x 256
    int w = caps_s[b * (TT + 1)];
    emb[b * EMB + t] = embd[(size_t)w * EMB + t];
}

// GRU gating: out = (1-z)*n + z*h   (gi/gh are [B][3*DD])
__global__ void k_gru_gate(const float* __restrict__ gi, const float* __restrict__ gh,
                           const float* __restrict__ h, float* __restrict__ out) {
    int i = blockIdx.x * 256 + threadIdx.x;          // BB*DD
    int b = i >> 9, j = i & (DD - 1);
    const float* a = gi + (size_t)b * 3 * DD;
    const float* c = gh + (size_t)b * 3 * DD;
    float r = 1.f / (1.f + __expf(-(a[j] + c[j])));
    float z = 1.f / (1.f + __expf(-(a[DD + j] + c[DD + j])));
    float n = tanhf(a[2 * DD + j] + r * c[2 * DD + j]);
    out[i] = (1.f - z) * n + z * h[i];
}

// score[b][p] = sum_a tanh(att1+att2+att3) * faw[a] + fab   (one wave per row)
__global__ __launch_bounds__(256) void k_score(
    const float* __restrict__ att1, const float* __restrict__ att2,
    const float* __restrict__ att3, const float* __restrict__ faw,
    const float* __restrict__ fab, float* __restrict__ score) {
    int wave = threadIdx.x >> 5, lane = threadIdx.x & 31;
    int row = blockIdx.x * 8 + wave;                 // BB*PP rows
    int b = row / PP;
    const float* a1 = att1 + (size_t)row * AD;
    const float* a2 = att2 + (size_t)b * AD;
    const float* a3 = att3 + (size_t)b * AD;
    float s = 0.f;
    for (int j = lane; j < AD; j += 32)
        s += tanhf(a1[j] + a2[j] + a3[j]) * faw[j];
    for (int off = 16; off > 0; off >>= 1) s += __shfl_xor(s, off, 32);
    if (lane == 0) score[row] = s + fab[0];
}

// softmax over P, cv += alpha, masked alpha output, context = alpha-weighted enc
__global__ __launch_bounds__(256) void k_softmax_ctx(
    const float* __restrict__ score, const float* __restrict__ enc,
    const int* __restrict__ si, const int* __restrict__ lens_s,
    float* __restrict__ cv, float* __restrict__ ctx,
    float* __restrict__ alphas_out, int t) {
    int b = blockIdx.x, tid = threadIdx.x;
    __shared__ float al[PP];
    __shared__ float red[256];
    float m = -INFINITY;
    for (int p = tid; p < PP; p += 256) m = fmaxf(m, score[b * PP + p]);
    red[tid] = m; __syncthreads();
    for (int s = 128; s > 0; s >>= 1) { if (tid < s) red[tid] = fmaxf(red[tid], red[tid + s]); __syncthreads(); }
    float mx = red[0]; __syncthreads();
    float sum = 0.f;
    for (int p = tid; p < PP; p += 256) { float e = __expf(score[b * PP + p] - mx); al[p] = e; sum += e; }
    red[tid] = sum; __syncthreads();
    for (int s = 128; s > 0; s >>= 1) { if (tid < s) red[tid] += red[tid + s]; __syncthreads(); }
    float inv = 1.f / red[0]; __syncthreads();
    bool active = (t == 0) || (t < lens_s[b]);
    for (int p = tid; p < PP; p += 256) {
        float a = al[p] * inv; al[p] = a;
        cv[b * PPAD + p] += a;
        alphas_out[((size_t)b * TT + t) * PP + p] = active ? a : 0.f;
    }
    __syncthreads();
    const float* e = enc + (size_t)si[b] * PP * ED;
    for (int d = tid; d < ED; d += 256) {
        float s2 = 0.f;
        for (int p = 0; p < PP; ++p) s2 += al[p] * e[(size_t)p * ED + d];
        ctx[b * ED + d] = s2;
    }
}

// masked write of predictions to d_out
__global__ void k_write_preds(const float* __restrict__ preds, const int* __restrict__ lens_s,
                              float* __restrict__ dout, int t) {
    int i = blockIdx.x * 256 + threadIdx.x;          // BB*VV
    int b = i / VV, v = i - b * VV;
    bool active = (t == 0) || (t < lens_s[b]);
    dout[((size_t)b * TT + t) * VV + v] = active ? preds[i] : 0.f;
}

// row argmax (first-index ties) + next embedding
__global__ __launch_bounds__(256) void k_argmax_embed(
    const float* __restrict__ preds, const float* __restrict__ embd,
    float* __restrict__ emb_out) {
    int b = blockIdx.x, tid = threadIdx.x;
    float bv = -INFINITY; int bi = 0x7fffffff;
    for (int i = tid; i < VV; i += 256) {
        float v = preds[(size_t)b * VV + i];
        if (v > bv) { bv = v; bi = i; }
    }
    __shared__ float sv[256]; __shared__ int sidx[256];
    sv[tid] = bv; sidx[tid] = bi; __syncthreads();
    for (int s = 128; s > 0; s >>= 1) {
        if (tid < s) {
            float ov = sv[tid + s]; int oi = sidx[tid + s];
            if (ov > sv[tid] || (ov == sv[tid] && oi < sidx[tid])) { sv[tid] = ov; sidx[tid] = oi; }
        }
        __syncthreads();
    }
    int w = sidx[0];
    emb_out[b * EMB + tid] = embd[(size_t)w * EMB + tid];
}

// ---------------- host orchestration ----------------
extern "C" void kernel_launch(void* const* d_in, const int* in_sizes, int n_in,
                              void* d_out, int out_size, void* d_ws, size_t ws_size,
                              hipStream_t stream) {
    (void)in_sizes; (void)n_in; (void)out_size; (void)ws_size;
    const float* enc       = (const float*)d_in[0];
    const int*   caps      = (const int*)d_in[1];
    const int*   clens     = (const int*)d_in[2];
    const float* embedding = (const float*)d_in[3];
    const float* g1_wih = (const float*)d_in[4];
    const float* g1_whh = (const float*)d_in[5];
    const float* g1_bih = (const float*)d_in[6];
    const float* g1_bhh = (const float*)d_in[7];
    const float* g2_wih = (const float*)d_in[8];
    const float* g2_whh = (const float*)d_in[9];
    const float* g2_bih = (const float*)d_in[10];
    const float* g2_bhh = (const float*)d_in[11];
    const float* enca_w = (const float*)d_in[12];
    const float* enca_b = (const float*)d_in[13];
    const float* deca_w = (const float*)d_in[14];
    const float* deca_b = (const float*)d_in[15];
    const float* conv_w = (const float*)d_in[16];
    const float* conv_b = (const float*)d_in[17];
    const float* cona_w = (const float*)d_in[18];
    const float* cona_b = (const float*)d_in[19];
    const float* faw    = (const float*)d_in[20];
    const float* fab    = (const float*)d_in[21];
    const float* s_w    = (const float*)d_in[22];
    const float* s_b    = (const float*)d_in[23];
    const float* fc_w   = (const float*)d_in[24];
    const float* fc_b   = (const float*)d_in[25];

    float* out = (float*)d_out;
    float* out_preds   = out;
    float* out_caps    = out + (size_t)BB * TT * VV;
    float* out_lens    = out_caps + (size_t)BB * (TT + 1);
    float* out_alphas  = out_lens + BB;
    float* out_sortind = out_alphas + (size_t)BB * TT * PP;

    char* ws = (char*)d_ws;
    size_t off = 0;
    auto alloc = [&](size_t bytes) -> void* {
        void* p = ws + off;
        off += (bytes + 255) & ~(size_t)255;
        return p;
    };
    int* sort_ind = (int*)alloc(BB * 4);
    int* lens_s   = (int*)alloc(BB * 4);
    int* caps_s   = (int*)alloc(BB * (TT + 1) * 4);
    unsigned short* fcw_bf  = (unsigned short*)alloc((size_t)VV * DD * 2);
    unsigned short* g1ih_bf = (unsigned short*)alloc((size_t)3 * DD * EMB * 2);
    unsigned short* g1hh_bf = (unsigned short*)alloc((size_t)3 * DD * DD * 2);
    unsigned short* g2ih_bf = (unsigned short*)alloc((size_t)3 * DD * ED * 2);
    unsigned short* g2hh_bf = (unsigned short*)alloc((size_t)3 * DD * DD * 2);
    unsigned short* deca_bf = (unsigned short*)alloc((size_t)AD * DD * 2);
    unsigned short* convw_bf= (unsigned short*)alloc((size_t)CD * PPAD * 2);
    unsigned short* cona_bf = (unsigned short*)alloc((size_t)AD * CD * 2);
    unsigned short* enca_bf = (unsigned short*)alloc((size_t)AD * ED * 2);
    unsigned short* sw_bf   = (unsigned short*)alloc((size_t)DD * ED * 2);
    unsigned short* enc_bf  = (unsigned short*)alloc((size_t)BB * PP * ED * 2);
    float* att1   = (float*)alloc((size_t)BB * PP * AD * 4);
    float* preds  = (float*)alloc((size_t)BB * VV * 4);
    float* sbuf   = (float*)alloc((size_t)BB * DD * 4);
    float* st_hat = (float*)alloc((size_t)BB * DD * 4);
    float* ctx    = (float*)alloc((size_t)BB * ED * 4);
    float* emb    = (float*)alloc((size_t)BB * EMB * 4);
    float* att2   = (float*)alloc((size_t)BB * AD * 4);
    float* att3   = (float*)alloc((size_t)BB * AD * 4);
    float* cbuf   = (float*)alloc((size_t)BB * CD * 4);
    float* cv     = (float*)alloc((size_t)BB * PPAD * 4);
    float* score  = (float*)alloc((size_t)BB * PP * 4);
    float* gi     = (float*)alloc((size_t)BB * 3 * DD * 4);
    float* gh     = (float*)alloc((size_t)BB * 3 * DD * 4);
    float* meanenc= (float*)alloc((size_t)BB * ED * 4);
    unsigned short* emb_bf   = (unsigned short*)alloc((size_t)BB * EMB * 2);
    unsigned short* s_bf     = (unsigned short*)alloc((size_t)BB * DD * 2);
    unsigned short* cv_bf    = (unsigned short*)alloc((size_t)BB * PPAD * 2);
    unsigned short* c_bf     = (unsigned short*)alloc((size_t)BB * CD * 2);
    unsigned short* ctx_bf   = (unsigned short*)alloc((size_t)BB * ED * 2);
    unsigned short* sthat_bf = (unsigned short*)alloc((size_t)BB * DD * 2);
    unsigned short* st_bf    = (unsigned short*)alloc((size_t)BB * DD * 2);
    unsigned short* mean_bf  = (unsigned short*)alloc((size_t)BB * ED * 2);

    auto cvt = [&](const float* s, unsigned short* d, int n) {
        k_f2bf<<<(n + 255) / 256, 256, 0, stream>>>(s, d, n);
    };
    // M=64 strip GEMM with async-LDS A staging
    auto gemm64 = [&](const unsigned short* A, const unsigned short* Bm, const float* bias,
                      float* C, int N, int K) {
        int blocks = ((N >> 4) + 7) / 8;
        size_t lds = (size_t)64 * (K + 8) * 2;
        gemm_bf16_strip<<<blocks, 256, lds, stream>>>(A, Bm, bias, C, N, K);
    };

    // ---- one-time setup ----
    k_sort<<<1, BB, 0, stream>>>(clens, caps, sort_ind, lens_s, caps_s,
                                 out_caps, out_lens, out_sortind);
    cvt(fc_w,   fcw_bf,  VV * DD);
    cvt(g1_wih, g1ih_bf, 3 * DD * EMB);
    cvt(g1_whh, g1hh_bf, 3 * DD * DD);
    cvt(g2_wih, g2ih_bf, 3 * DD * ED);
    cvt(g2_whh, g2hh_bf, 3 * DD * DD);
    cvt(deca_w, deca_bf, AD * DD);
    cvt(cona_w, cona_bf, AD * CD);
    cvt(enca_w, enca_bf, AD * ED);
    cvt(s_w,    sw_bf,   DD * ED);
    k_convw_bf<<<(CD * PPAD) / 256, 256, 0, stream>>>(conv_w, convw_bf);

    k_gather_enc_bf<<<(BB * PP * ED) / 256, 256, 0, stream>>>(enc, sort_ind, enc_bf);
    {   // att1 (25600x256, K=512): per-tile kernel, plenty of parallelism
        int tiles = ((BB * PP) >> 4) * (AD >> 4);
        gemm_bf16_nt<<<(tiles + 7) / 8, 256, 0, stream>>>(enc_bf, enca_bf, enca_b,
                                                          att1, BB * PP, AD, ED);
    }

    k_mean_enc<<<(BB * ED) / 256, 256, 0, stream>>>(enc, sort_ind, meanenc);
    cvt(meanenc, mean_bf, BB * ED);
    gemm64(mean_bf, sw_bf, s_b, sbuf, DD, ED);                  // s0

    k_embed0<<<BB, EMB, 0, stream>>>(caps_s, embedding, emb);   // emb0
    k_zero<<<(BB * PPAD + 255) / 256, 256, 0, stream>>>(cv, BB * PPAD);
    k_zero<<<(BB * AD + 255) / 256, 256, 0, stream>>>(att3, BB * AD);

    // ---- 32 sequential decode steps ----
    for (int t = 0; t < TT; ++t) {
        cvt(emb,  emb_bf, BB * EMB);
        cvt(sbuf, s_bf,   BB * DD);
        gemm64(emb_bf, g1ih_bf, g1_bih, gi, 3 * DD, EMB);       // gi1
        gemm64(s_bf,   g1hh_bf, g1_bhh, gh, 3 * DD, DD);        // gh1
        k_gru_gate<<<(BB * DD) / 256, 256, 0, stream>>>(gi, gh, sbuf, st_hat);

        gemm64(s_bf, deca_bf, deca_b, att2, AD, DD);            // att2 from old s
        if (t > 0) {
            cvt(cv, cv_bf, BB * PPAD);
            gemm64(cv_bf, convw_bf, conv_b, cbuf, CD, PPAD);    // c
            cvt(cbuf, c_bf, BB * CD);
            gemm64(c_bf, cona_bf, cona_b, att3, AD, CD);        // att3
        }
        k_score<<<(BB * PP) / 8, 256, 0, stream>>>(att1, att2, att3, faw, fab, score);
        k_softmax_ctx<<<BB, 256, 0, stream>>>(score, enc, sort_ind, lens_s,
                                              cv, ctx, out_alphas, t);

        cvt(ctx,    ctx_bf,   BB * ED);
        cvt(st_hat, sthat_bf, BB * DD);
        gemm64(ctx_bf,   g2ih_bf, g2_bih, gi, 3 * DD, ED);      // gi2
        gemm64(sthat_bf, g2hh_bf, g2_bhh, gh, 3 * DD, DD);      // gh2
        k_gru_gate<<<(BB * DD) / 256, 256, 0, stream>>>(gi, gh, st_hat, sbuf); // new state

        cvt(sbuf, st_bf, BB * DD);
        gemm64(st_bf, fcw_bf, fc_b, preds, VV, DD);             // FC vocab projection
        k_write_preds<<<(BB * VV) / 256, 256, 0, stream>>>(preds, lens_s, out_preds, t);
        k_argmax_embed<<<BB, 256, 0, stream>>>(preds, embedding, emb);
    }
}
